// SNN_MLP_8658654069278
// MI455X (gfx1250) — matrix operand order repaired and checked by reading, not compiled, not run on hardware
//
#include <hip/hip_runtime.h>
#include <hip/hip_bf16.h>

typedef __attribute__((ext_vector_type(16))) _Float16 v16h;
typedef __attribute__((ext_vector_type(8)))  _Float16 v8h;
typedef __attribute__((ext_vector_type(8)))  float    v8f;

#define T_STEPS 1024
#define B_TOT   128
#define C1      256   // conv out channels / GEMM1 K
#define H1      512   // hidden / GEMM1 N / GEMM2 K
#define NC      10    // classes
#define NCP     16    // padded classes (one WMMA n-tile)
#define XLEN    2048

// ---------------- prep: transpose + f16-convert weights into workspace ----
// w1  [K=256, N=512] f32  ->  w1t [N=512, K=256] f16   (N-major rows)
__global__ void snn_prep_w1(const float* __restrict__ w1, _Float16* __restrict__ w1t) {
    int i = blockIdx.x * blockDim.x + threadIdx.x;
    if (i < H1 * C1) {
        int n = i >> 8;          // 0..511
        int k = i & 255;         // 0..255
        w1t[i] = (_Float16)w1[k * H1 + n];
    }
}
// w2  [K=512, N=10] f32  ->  w2t [N=16, K=512] f16, zero padded cols 10..15
__global__ void snn_prep_w2(const float* __restrict__ w2, _Float16* __restrict__ w2t) {
    int i = blockIdx.x * blockDim.x + threadIdx.x;
    if (i < NCP * H1) {
        int n = i >> 9;          // 0..15
        int k = i & 511;         // 0..511
        w2t[i] = (n < NC) ? (_Float16)w2[k * NC + n] : (_Float16)0.0f;
    }
}

// ---------------- persistent SNN scan kernel ------------------------------
// grid = 8 workgroups (one per 16-row batch tile), block = 1024 threads (32 waves)
__global__ __launch_bounds__(1024, 1)
void snn_scan_kernel(const float* __restrict__ x,       // [128, 2048]
                     const float* __restrict__ wconv,   // [256, 1, 4]
                     const _Float16* __restrict__ w1t,  // [512, 256] f16
                     const _Float16* __restrict__ w2t,  // [16, 512]  f16
                     float* __restrict__ out)           // [128, 10]
{
    // Weights resident in LDS for the whole scan (WGP has 320 KB).
    __shared__ _Float16 sm_w1[H1 * C1];   // 256 KB, [n=512][k=256]
    __shared__ _Float16 sm_w2[NCP * H1];  // 16 KB,  [n=16][k=512]
    __shared__ _Float16 sm_s1[16 * C1];   // 8 KB,   spikes layer1 [m][k]
    __shared__ _Float16 sm_s2[16 * H1];   // 16 KB,  spikes layer2 [m][k]
    __shared__ float    sm_xw[16 * 4];    // conv input window [b_local][k]

    const int tid    = threadIdx.x;
    const int lane   = tid & 31;
    const int wid    = tid >> 5;          // 0..31
    const int b_base = blockIdx.x * 16;

    // ---- stage weights global -> LDS (vectorized b128 copies) ------------
    {
        const v8h* src1 = (const v8h*)w1t;
        v8h*       dst1 = (v8h*)sm_w1;
        #pragma unroll
        for (int r = 0; r < (H1 * C1 / 8) / 1024; ++r)
            dst1[r * 1024 + tid] = src1[r * 1024 + tid];
        const v8h* src2 = (const v8h*)w2t;
        v8h*       dst2 = (v8h*)sm_w2;
        if (tid < NCP * H1 / 8) dst2[tid] = src2[tid];
    }

    // ---- conv/LIF1 ownership: c = tid&255, 4 batch rows per thread -------
    const int c    = tid & 255;
    const int bloc = tid >> 8;            // 0..3 -> b_local = bloc*4 + j
    const float wc0 = wconv[c * 4 + 0];
    const float wc1 = wconv[c * 4 + 1];
    const float wc2 = wconv[c * 4 + 2];
    const float wc3 = wconv[c * 4 + 3];
    float v1r[4] = {0.f, 0.f, 0.f, 0.f};

    // ---- persistent state in WMMA C/D layout ------------------------------
    v8f v2 = {};                          // membrane, this wave's 16x16 tile of [16,512]
    v8f v3 = {};                          // wave 0 only: [16,16] tile of output layer
    v8f outacc = {};                      // wave 0 only: spike sum over time

    // fragment addressing (contiguous k = 16*(lane/16) + s for both A and B)
    const int a_row  = lane & 15;                   // A: lane -> matrix row
    const int a_klo  = (lane >> 4) ? 16 : 0;        // A: K offset for hi lanes
    const int b_col  = lane & 15;                   // B: lane -> matrix col
    const int b_koff = (lane >> 4) ? 16 : 0;        // B: K offset for hi lanes
    const int nt     = wid;                         // GEMM1 n-tile of this wave

    __syncthreads();                      // weights staged

    for (int t = 0; t < T_STEPS; ++t) {
        // Opaque per-iteration zero: stops the compiler from hoisting the
        // loop-invariant LDS weight-fragment loads into (spilling) registers.
        uint32_t tz;
        asm volatile("s_mov_b32 %0, 0" : "=s"(tz));
        const _Float16* w1l = sm_w1 + tz;
        const _Float16* w2l = sm_w2 + tz;

        // ---- phase A: stage conv input window (x[b, 2t-1 .. 2t+2]) -------
        if (tid < 64) {
            int bl  = tid >> 2;
            int k   = tid & 3;
            int idx = 2 * t + k - 1;
            float v = 0.0f;
            if (idx >= 0 && idx < XLEN) v = x[(b_base + bl) * XLEN + idx];
            sm_xw[bl * 4 + k] = v;
        }
        __syncthreads();

        // ---- phase B: conv1d + LIF1 -> s1 spikes in LDS ------------------
        #pragma unroll
        for (int j = 0; j < 4; ++j) {
            int bl = bloc * 4 + j;
            float y = sm_xw[bl * 4 + 0] * wc0 + sm_xw[bl * 4 + 1] * wc1 +
                      sm_xw[bl * 4 + 2] * wc2 + sm_xw[bl * 4 + 3] * wc3;
            float v = v1r[j] * 0.5f + y;            // v = v*(1-1/tau) + x
            float s = (v >= 1.0f) ? 1.0f : 0.0f;    // H(v - vth)
            v1r[j]  = v - s;                        // soft reset
            sm_s1[bl * C1 + c] = (_Float16)s;
        }
        __syncthreads();

        // ---- phase C: GEMM1 (s1 @ w1) + LIF2 -> s2 spikes ----------------
        {
            v8f acc = {};
            #pragma unroll
            for (int kt = 0; kt < 8; ++kt) {
                // A fragment: 16x32 f16 from LDS s1
                const _Float16* ap = &sm_s1[a_row * C1 + kt * 32 + a_klo];
                v8h alo = *(const v8h*)(ap);
                v8h ahi = *(const v8h*)(ap + 8);
                v16h a;
                #pragma unroll
                for (int j = 0; j < 8; ++j) { a[j] = alo[j]; a[8 + j] = ahi[j]; }
                // B fragment: 32x16 f16, contiguous N-major row in LDS
                v16h b = *(const v16h*)(w1l + (nt * 16 + b_col) * C1 + kt * 32 + b_koff);
                acc = __builtin_amdgcn_wmma_f32_16x16x32_f16(
                        false, a, false, b, (short)0, acc, false, false);
            }
            // LIF2 elementwise on accumulator-resident membrane state
            #pragma unroll
            for (int i = 0; i < 8; ++i) {
                float v = v2[i] * 0.5f + acc[i];
                float s = (v >= 1.0f) ? 1.0f : 0.0f;
                v2[i]   = v - s;
                int row = i + ((lane >> 4) << 3);   // C/D layout: vgpr i, lane half
                sm_s2[row * H1 + nt * 16 + (lane & 15)] = (_Float16)s;
            }
        }
        __syncthreads();

        // ---- phase D: GEMM2 (s2 @ w2) + LIF3 (wave 0 only) ---------------
        if (wid == 0) {
            v8f acc = {};
            #pragma unroll
            for (int kt = 0; kt < 16; ++kt) {
                const _Float16* ap = &sm_s2[a_row * H1 + kt * 32 + a_klo];
                v8h alo = *(const v8h*)(ap);
                v8h ahi = *(const v8h*)(ap + 8);
                v16h a;
                #pragma unroll
                for (int j = 0; j < 8; ++j) { a[j] = alo[j]; a[8 + j] = ahi[j]; }
                v16h b = *(const v16h*)(w2l + b_col * H1 + kt * 32 + b_koff);
                acc = __builtin_amdgcn_wmma_f32_16x16x32_f16(
                        false, a, false, b, (short)0, acc, false, false);
            }
            #pragma unroll
            for (int i = 0; i < 8; ++i) {
                float v = v3[i] * 0.5f + acc[i];
                float s = (v >= 1.0f) ? 1.0f : 0.0f;
                v3[i]      = v - s;
                outacc[i] += s;
            }
        }
        // no barrier needed here: next-iter LDS writes don't alias phase-D reads
    }

    // ---- epilogue: mean over T, write [16,10] tile -----------------------
    if (wid == 0) {
        int col = lane & 15;
        if (col < NC) {
            #pragma unroll
            for (int i = 0; i < 8; ++i) {
                int row = i + ((lane >> 4) << 3);
                out[(b_base + row) * NC + col] = outacc[i] * (1.0f / (float)T_STEPS);
            }
        }
    }
}

extern "C" void kernel_launch(void* const* d_in, const int* in_sizes, int n_in,
                              void* d_out, int out_size, void* d_ws, size_t ws_size,
                              hipStream_t stream) {
    const float* x     = (const float*)d_in[0];   // [128, 2048]
    const float* wconv = (const float*)d_in[1];   // [256, 1, 4]
    const float* w1    = (const float*)d_in[2];   // [256, 512]
    const float* w2    = (const float*)d_in[3];   // [512, 10]
    float* out = (float*)d_out;                   // [128, 10]

    _Float16* w1t = (_Float16*)d_ws;                                   // 256 KB
    _Float16* w2t = (_Float16*)((char*)d_ws + (size_t)H1 * C1 * 2);    // 16 KB

    snn_prep_w1<<<(H1 * C1 + 255) / 256, 256, 0, stream>>>(w1, w1t);
    snn_prep_w2<<<(NCP * H1 + 255) / 256, 256, 0, stream>>>(w2, w2t);
    snn_scan_kernel<<<dim3(B_TOT / 16), dim3(1024), 0, stream>>>(x, wconv, w1t, w2t, out);
}
